// VanillaRNN_50208167690394
// MI455X (gfx1250) — matrix operand order repaired and checked by reading, not compiled, not run on hardware
//
#include <hip/hip_runtime.h>
#include <hip/hip_bf16.h>
#include <math.h>

// Problem constants from the reference
#define T_STEPS 129
#define BB 1024
#define II 512
#define HH 2048
#define KSPLIT ((T_STEPS - 3) / 2)      // 63
#define ENC_STEPS (KSPLIT + 2)          // 65
#define DEC_STEPS (KSPLIT + 1)          // 64

typedef __bf16 bf16_t;
typedef __attribute__((ext_vector_type(16))) __bf16 v16bf;
typedef __attribute__((ext_vector_type(8)))  __bf16 v8bf;
typedef __attribute__((ext_vector_type(8)))  float  v8f;

// ---------------------------------------------------------------------------
// Fragment loaders per CDNA5 WMMA VGPR layouts (ISA 7.12.2, wave32).
// LDA/LDB are compile-time so all stride math is shifts + immediate offsets.
// ---------------------------------------------------------------------------

// A (16x32 bf16, MxK): lanes 0-15 -> rows 0-15 with K = 0..7 (v0..v3) and
// 16..23 (v4..v7); lanes 16-31 -> same rows with K = 8..15 and 24..31.
template <int LDA>
__device__ __forceinline__ v16bf load_a_frag(const bf16_t* __restrict__ A,
                                             int rowBase, int k0, int lane) {
  int half = lane >> 4;      // 0 or 1
  int r    = lane & 15;
  const bf16_t* p = A + (size_t)(rowBase + r) * LDA + k0 + half * 8;
  v8bf lo = *(const v8bf*)(p);        // K = half*8 + 0..7
  v8bf hi = *(const v8bf*)(p + 16);   // K = 16 + half*8 + 0..7
  v16bf f;
#pragma unroll
  for (int i = 0; i < 8; ++i) { f[i] = lo[i]; f[8 + i] = hi[i]; }
  return f;
}

// B (32x16 bf16, KxN) with B(k,n) = W[n][k], W row-major over k:
// lanes 0-15 -> col n = lane, K = 0..7 and 16..23; lanes 16-31 -> col
// n = lane-16, K = 8..15 and 24..31 (same K order as the A fragment).
template <int LDB>
__device__ __forceinline__ v16bf load_b_frag(const bf16_t* __restrict__ W,
                                             int colBase, int k0, int lane) {
  int cHalf = lane >> 4;
  int c     = lane & 15;
  const bf16_t* p = W + (size_t)(colBase + c) * LDB + k0 + cHalf * 8;
  v8bf lo = *(const v8bf*)(p);
  v8bf hi = *(const v8bf*)(p + 16);
  v16bf f;
#pragma unroll
  for (int i = 0; i < 8; ++i) { f[i] = lo[i]; f[8 + i] = hi[i]; }
  return f;
}

// ---------------------------------------------------------------------------
// WMMA GEMM: D[m][n] = act( bias[n] + sum_k A(m,k) * W[n][k] )
// A split into A0 (K0 cols, may be 0) and A1 (K1 cols); W is NN x (K0+K1).
// Wave tile: 32(M) x 16*NT(N); block = 8 waves (2 M x 4 N) -> 64 x 64*NT.
// TANH: apply tanh and store bf16 (next hidden state), else raw f32 (logits).
// ---------------------------------------------------------------------------
template <int K0, int K1, int NN, int NT, bool TANH>
__global__ __launch_bounds__(256, 2)
void wmma_gemm_kernel(const bf16_t* __restrict__ A0,
                      const bf16_t* __restrict__ A1,
                      const bf16_t* __restrict__ W,   // NN x (K0+K1)
                      const float* __restrict__ bias, // NN
                      void* __restrict__ Dout) {
  constexpr int KTOT = K0 + K1;
  const int tid    = threadIdx.x;
  const int lane   = tid & 31;
  const int waveId = tid >> 5;
  const int wy     = waveId >> 2;                  // 0..1  (M)
  const int wx     = waveId & 3;                   // 0..3  (N)
  const int waveM  = blockIdx.y * 64 + wy * 32;
  const int waveN  = (blockIdx.x * 4 + wx) * (16 * NT);

  v8f acc[2][NT];
#pragma unroll
  for (int i = 0; i < 2; ++i)
#pragma unroll
    for (int j = 0; j < NT; ++j) acc[i][j] = (v8f){};

  // ---- K-loop part 1: x-projection (compiled away when K0 == 0) ----
  if (K0 > 0) {
#pragma unroll 2
    for (int k0 = 0; k0 < K0; k0 += 32) {
      v16bf a0 = load_a_frag<K0>(A0, waveM,      k0, lane);
      v16bf a1 = load_a_frag<K0>(A0, waveM + 16, k0, lane);
#pragma unroll
      for (int j = 0; j < NT; ++j) {
        v16bf b = load_b_frag<KTOT>(W, waveN + j * 16, k0, lane);
        acc[0][j] = __builtin_amdgcn_wmma_f32_16x16x32_bf16(
            false, a0, false, b, (short)0, acc[0][j], false, false);
        acc[1][j] = __builtin_amdgcn_wmma_f32_16x16x32_bf16(
            false, a1, false, b, (short)0, acc[1][j], false, false);
      }
    }
  }

  // ---- K-loop part 2: recurrent h part ----
#pragma unroll 2
  for (int k0 = 0; k0 < K1; k0 += 32) {
    v16bf a0 = load_a_frag<K1>(A1, waveM,      k0, lane);
    v16bf a1 = load_a_frag<K1>(A1, waveM + 16, k0, lane);
#pragma unroll
    for (int j = 0; j < NT; ++j) {
      v16bf b = load_b_frag<KTOT>(W, waveN + j * 16, K0 + k0, lane);
      acc[0][j] = __builtin_amdgcn_wmma_f32_16x16x32_bf16(
          false, a0, false, b, (short)0, acc[0][j], false, false);
      acc[1][j] = __builtin_amdgcn_wmma_f32_16x16x32_bf16(
          false, a1, false, b, (short)0, acc[1][j], false, false);
    }
  }

  // C/D layout: lane l holds column n = (l&15); v8f element r -> row
  // M = r + (l>=16 ? 8 : 0).
  const int cHalf = lane >> 4;
  const int c     = lane & 15;
  bf16_t* obf = (bf16_t*)Dout;
  float*  of  = (float*)Dout;

#pragma unroll
  for (int i = 0; i < 2; ++i) {
#pragma unroll
    for (int j = 0; j < NT; ++j) {
      int colg = waveN + j * 16 + c;
      float bv = bias[colg];
#pragma unroll
      for (int r = 0; r < 8; ++r) {
        int rowg = waveM + i * 16 + cHalf * 8 + r;
        float v = acc[i][j][r] + bv;
        if (TANH) {
          obf[(size_t)rowg * NN + colg] = (bf16_t)tanhf(v);
        } else {
          of[(size_t)rowg * NN + colg] = v;
        }
      }
    }
  }
}

// ---------------------------------------------------------------------------
// Decoder output: log_softmax over 512 logits, argmax==2 -> done gating,
// write gated output row. One block (256 threads) per batch row.
// ---------------------------------------------------------------------------
__global__ __launch_bounds__(256)
void dec_output_kernel(const float* __restrict__ logits,  // BB x II
                       float* __restrict__ done,          // BB
                       float* __restrict__ outp) {        // BB x II
  __shared__ float sval[256];
  __shared__ int   sidx[256];
  __shared__ float sred[256];
  __shared__ float s_b, s_lse;

  const int row = blockIdx.x;
  const int tid = threadIdx.x;
  const float* lr = logits + (size_t)row * II;

  float v0 = lr[tid];
  float v1 = lr[tid + 256];

  float mv; int mi;
  if (v1 > v0) { mv = v1; mi = tid + 256; } else { mv = v0; mi = tid; }
  sval[tid] = mv; sidx[tid] = mi;
  __syncthreads();
#pragma unroll
  for (int s = 128; s > 0; s >>= 1) {
    if (tid < s) {
      float ov = sval[tid + s]; int oi = sidx[tid + s];
      if (ov > sval[tid] || (ov == sval[tid] && oi < sidx[tid])) {
        sval[tid] = ov; sidx[tid] = oi;
      }
    }
    __syncthreads();
  }
  float m   = sval[0];
  int amax  = sidx[0];

  float e = expf(v0 - m) + expf(v1 - m);
  sred[tid] = e;
  __syncthreads();
#pragma unroll
  for (int s = 128; s > 0; s >>= 1) {
    if (tid < s) sred[tid] += sred[tid + s];
    __syncthreads();
  }

  if (tid == 0) {
    float lse = m + logf(sred[0]);
    float z   = (amax == 2) ? 1.0f : 0.0f;
    float dn  = fmaxf(done[row], z);
    done[row] = dn;
    s_b = dn; s_lse = lse;
  }
  __syncthreads();
  float b = s_b, lse = s_lse;

  float o0 = (1.0f - b) * (v0 - lse) + b * ((tid == 0) ? 1.0f : 0.0f);
  float o1 = (1.0f - b) * (v1 - lse);   // index tid+256 is never 0
  outp[(size_t)row * II + tid]       = o0;
  outp[(size_t)row * II + tid + 256] = o1;
}

// ---------------------------------------------------------------------------
// Setup helpers
// ---------------------------------------------------------------------------
__global__ void f32_to_bf16_kernel(const float* __restrict__ in,
                                   bf16_t* __restrict__ out, int n) {
  int i = blockIdx.x * blockDim.x + threadIdx.x;
  if (i < n) out[i] = (bf16_t)in[i];
}

// Pack [Wih | Whh] row-major into a single N x (K0+K1) bf16 matrix.
__global__ void pack_enc_weights_kernel(const float* __restrict__ Wih,
                                        const float* __restrict__ Whh,
                                        bf16_t* __restrict__ out,
                                        int N, int K0, int K1) {
  int idx = blockIdx.x * blockDim.x + threadIdx.x;
  int K = K0 + K1;
  if (idx < N * K) {
    int n = idx / K, k = idx - n * K;
    float v = (k < K0) ? Wih[(size_t)n * K0 + k]
                       : Whh[(size_t)n * K1 + (k - K0)];
    out[idx] = (bf16_t)v;
  }
}

__global__ void bias_sum_kernel(const float* __restrict__ a,
                                const float* __restrict__ b,
                                float* __restrict__ o, int n) {
  int i = blockIdx.x * blockDim.x + threadIdx.x;
  if (i < n) o[i] = a[i] + b[i];
}

__global__ void zero_bf16_kernel(bf16_t* p, int n) {
  int i = blockIdx.x * blockDim.x + threadIdx.x;
  if (i < n) p[i] = (bf16_t)0.0f;
}

__global__ void zero_f32_kernel(float* p, int n) {
  int i = blockIdx.x * blockDim.x + threadIdx.x;
  if (i < n) p[i] = 0.0f;
}

// ---------------------------------------------------------------------------
// Host-side launch
// ---------------------------------------------------------------------------
extern "C" void kernel_launch(void* const* d_in, const int* in_sizes, int n_in,
                              void* d_out, int out_size, void* d_ws, size_t ws_size,
                              hipStream_t stream) {
  (void)in_sizes; (void)n_in; (void)out_size; (void)ws_size;

  const float* x        = (const float*)d_in[0];   // (T, B, I)
  const float* enc_Wih  = (const float*)d_in[1];   // (H, I)
  const float* enc_Whh  = (const float*)d_in[2];   // (H, H)
  const float* enc_bih  = (const float*)d_in[3];   // (H,)
  const float* enc_bhh  = (const float*)d_in[4];   // (H,)
  const float* dec_Whh  = (const float*)d_in[6];   // (H, H)   (dec_Wih unused)
  const float* dec_bih  = (const float*)d_in[7];   // (H,)
  const float* dec_bhh  = (const float*)d_in[8];   // (H,)
  const float* ro_W     = (const float*)d_in[9];   // (I, H)
  const float* ro_b     = (const float*)d_in[10];  // (I,)
  float* out = (float*)d_out;                      // (DEC_STEPS, B, I)

  // Workspace carve-up (256B aligned slices)
  char* w = (char*)d_ws;
  size_t off = 0;
  auto alloc = [&](size_t bytes) -> void* {
    void* p = (void*)(w + off);
    off = (off + bytes + 255) & ~(size_t)255;
    return p;
  };
  bf16_t* WencCat = (bf16_t*)alloc((size_t)HH * (II + HH) * 2); // 10.5 MB
  bf16_t* Wdec    = (bf16_t*)alloc((size_t)HH * HH * 2);        //  8.4 MB
  bf16_t* Wro     = (bf16_t*)alloc((size_t)II * HH * 2);        //  2.1 MB
  float*  encBias = (float*)alloc((size_t)HH * 4);
  float*  decBias = (float*)alloc((size_t)HH * 4);
  bf16_t* xbf     = (bf16_t*)alloc((size_t)BB * II * 2);        //  1.0 MB
  bf16_t* h0      = (bf16_t*)alloc((size_t)BB * HH * 2);        //  4.2 MB
  bf16_t* h1      = (bf16_t*)alloc((size_t)BB * HH * 2);        //  4.2 MB
  float*  logits  = (float*)alloc((size_t)BB * II * 4);         //  2.1 MB
  float*  done    = (float*)alloc((size_t)BB * 4);

  const int TPB = 256;
  auto blks = [](int n, int t) { return (n + t - 1) / t; };

  // ---- One-time (per call) setup: pack / convert weights, biases, state ----
  {
    int n = HH * (II + HH);
    pack_enc_weights_kernel<<<blks(n, TPB), TPB, 0, stream>>>(
        enc_Wih, enc_Whh, WencCat, HH, II, HH);
  }
  f32_to_bf16_kernel<<<blks(HH * HH, TPB), TPB, 0, stream>>>(dec_Whh, Wdec, HH * HH);
  f32_to_bf16_kernel<<<blks(II * HH, TPB), TPB, 0, stream>>>(ro_W, Wro, II * HH);
  bias_sum_kernel<<<blks(HH, TPB), TPB, 0, stream>>>(enc_bih, enc_bhh, encBias, HH);
  bias_sum_kernel<<<blks(HH, TPB), TPB, 0, stream>>>(dec_bih, dec_bhh, decBias, HH);
  zero_bf16_kernel<<<blks(BB * HH, TPB), TPB, 0, stream>>>(h0, BB * HH);
  zero_f32_kernel<<<blks(BB, TPB), TPB, 0, stream>>>(done, BB);

  bf16_t* hcur = h0;
  bf16_t* hnxt = h1;

  // Block tile: 64(M) x 64*NT(N). NT=4 -> 256-wide; NT=2 -> 128-wide.
  const dim3 gridEnc(HH / 256, BB / 64);   // (8,16)  enc step GEMM
  const dim3 gridDec(HH / 256, BB / 64);   // (8,16)  dec step GEMM
  const dim3 gridOut(II / 128, BB / 64);   // (4,16)  logits GEMM

  // ---- Encoder: 65 sequential fused [x_t | h] GEMM + tanh steps ----
  for (int t = 0; t < ENC_STEPS; ++t) {
    f32_to_bf16_kernel<<<blks(BB * II, TPB), TPB, 0, stream>>>(
        x + (size_t)t * BB * II, xbf, BB * II);
    wmma_gemm_kernel<II, HH, HH, 4, true><<<gridEnc, TPB, 0, stream>>>(
        xbf, hcur, WencCat, encBias, (void*)hnxt);
    bf16_t* tmp = hcur; hcur = hnxt; hnxt = tmp;
  }

  // ---- Decoder: 64 sequential steps ----
  for (int t = 0; t < DEC_STEPS; ++t) {
    wmma_gemm_kernel<0, HH, HH, 4, true><<<gridDec, TPB, 0, stream>>>(
        nullptr, hcur, Wdec, decBias, (void*)hnxt);
    wmma_gemm_kernel<0, HH, II, 2, false><<<gridOut, TPB, 0, stream>>>(
        nullptr, hnxt, Wro, ro_b, (void*)logits);
    dec_output_kernel<<<BB, TPB, 0, stream>>>(
        logits, done, out + (size_t)t * BB * II);
    bf16_t* tmp = hcur; hcur = hnxt; hnxt = tmp;
  }
}
